// Mamba2_6777458393768
// MI455X (gfx1250) — compile-verified, matrix operand order checked by reading, not run on hardware
//
#include <hip/hip_runtime.h>
#include <hip/hip_bf16.h>
#include <cstdint>

// ---------------- problem constants ----------------
#define SEQ      2048
#define DIM      2048
#define DIN      8352        // D_IN_PROJ = 2*4096 + 2*64 + 32
#define DI       4096        // DIM_INNER
#define CONVD    4224        // DIM_INNER + 2*D_STATE
#define NH       32
#define HD       128
#define DS       64
#define CHK      256
#define NC       8           // SEQ / CHUNK

typedef __bf16 bf16_t;
typedef __attribute__((ext_vector_type(16))) __bf16 v16bf;
typedef __attribute__((ext_vector_type(8)))  __bf16 v8bf;
typedef __attribute__((ext_vector_type(8)))  float  v8f;
typedef unsigned int v4u __attribute__((ext_vector_type(4)));
typedef int          v4i __attribute__((ext_vector_type(4)));
typedef int          v8i __attribute__((ext_vector_type(8)));

#define LDS_AS __attribute__((address_space(3)))

#if defined(__has_builtin)
#if __has_builtin(__builtin_amdgcn_tensor_load_to_lds)
#define HAVE_TDM 1
#endif
#endif
#ifndef HAVE_TDM
#define HAVE_TDM 0
#endif

// 32-bit LDS byte offset of a __shared__ pointer
__device__ __forceinline__ unsigned lds_off(const void* p) {
  return (unsigned)(size_t)(LDS_AS const void*)p;
}

// CDNA5 async DMA: global -> LDS, 16B per lane, tracked by ASYNCcnt
__device__ __forceinline__ void async_to_lds_b128(unsigned lds, const void* g) {
  asm volatile("global_load_async_to_lds_b128 %0, %1, off"
               :: "v"(lds), "v"(g) : "memory");
}

__device__ __forceinline__ void wait_async0() {
#if __has_builtin(__builtin_amdgcn_s_wait_asynccnt)
  __builtin_amdgcn_s_wait_asynccnt(0);
#else
  asm volatile("s_wait_asynccnt 0x0" ::: "memory");
#endif
}

#if HAVE_TDM
// TDM: DMA a 2-D tile (tile_h rows x tile_w elems, 2-byte elems) from a
// row-major global tensor (row stride = stride_elems) into tight row-major
// LDS at lds_byte_addr. D# layout per CDNA5 ISA 8.3/8.4. Tracked by TENSORcnt.
__device__ __forceinline__ void tdm_load_2d_bf16(unsigned lds_byte_addr,
                                                 const void* gaddr,
                                                 unsigned tensor_w,   // elems per row
                                                 unsigned tensor_h,   // rows
                                                 unsigned tile_w,     // elems
                                                 unsigned tile_h,     // rows
                                                 unsigned stride_elems) {
  unsigned long long ga = (unsigned long long)(size_t)gaddr;
  v4u g0;
  g0.x = 1u;                                                  // count=1 (valid), user
  g0.y = lds_byte_addr;                                       // lds_addr
  g0.z = (unsigned)ga;                                        // global_addr[31:0]
  g0.w = (unsigned)((ga >> 32) & 0x01FFFFFFu) | (2u << 30);   // addr[56:32] | type=2
  v8i g1;
  g1[0] = (int)(1u << 16);                                    // data_size=1 (2 bytes)
  g1[1] = (int)((tensor_w & 0xFFFFu) << 16);                  // tensor_dim0[15:0]
  g1[2] = (int)(((tensor_w >> 16) & 0xFFFFu) |                // tensor_dim0[31:16]
                ((tensor_h & 0xFFFFu) << 16));                // tensor_dim1[15:0]
  g1[3] = (int)(((tensor_h >> 16) & 0xFFFFu) |                // tensor_dim1[31:16]
                ((tile_w & 0xFFFFu) << 16));                  // tile_dim0
  g1[4] = (int)(tile_h & 0xFFFFu);                            // tile_dim1 (tile_dim2=0)
  g1[5] = (int)stride_elems;                                  // tensor_dim0_stride[31:0]
  g1[6] = 0;                                                  // stride hi, dim1_stride lo
  g1[7] = 0;
  v4i z4 = {0, 0, 0, 0};
#if __clang_major__ >= 23
  v8i z8 = {0, 0, 0, 0, 0, 0, 0, 0};
  __builtin_amdgcn_tensor_load_to_lds(g0, g1, z4, z4, z8, 0);
#else
  __builtin_amdgcn_tensor_load_to_lds(g0, g1, z4, z4, 0);
#endif
}
#endif

__device__ __forceinline__ void wait_stage0() {
#if HAVE_TDM
  __builtin_amdgcn_s_wait_tensorcnt(0);
#else
  wait_async0();
#endif
}

// ---------------- WMMA helpers ----------------
// A-fragment (16x32 row-major) and B-fragment (Kx16 from row-major (N,K))
// share per-lane addressing (ISA 7.12.2): lane L: row=L%16, half=L/16;
// elems 0..7 = K[half*8..+7], elems 8..15 = K[16+half*8..+7].
__device__ __forceinline__ v16bf load_frag(const bf16_t* base, int ld, int lane) {
  const int r    = lane & 15;
  const int half = lane >> 4;
  const bf16_t* p = base + (size_t)r * ld + half * 8;
  v16bf f;
  ((v8bf*)&f)[0] = *(const v8bf*)(p);
  ((v8bf*)&f)[1] = *(const v8bf*)(p + 16);
  return f;
}

__device__ __forceinline__ v8f wmma_bf16(v16bf a, v16bf b, v8f c) {
  return __builtin_amdgcn_wmma_f32_16x16x32_bf16(
      false, a, false, b, (short)0, c, false, false);
}

__device__ __forceinline__ float siluf(float x) {
  return x / (1.0f + __expf(-x));
}

// ---------------- fp32 -> bf16 convert ----------------
__global__ void k_cvt_bf16(const float* __restrict__ in, bf16_t* __restrict__ out, int n) {
  int i = blockIdx.x * 256 + threadIdx.x;
  if (i < n) out[i] = (bf16_t)in[i];
}

// ---------------- bf16 WMMA GEMM:  C[M,N] = A[M,K] * B[N,K]^T ----------------
// 256 threads = 8 waves; block tile 256(M) x 32(N); wave tile 32x32 (4 WMMA).
// Double-buffered LDS filled by TDM (tensor_load_to_lds) or async-to-LDS.
#define GTM 256
#define GTN 32
#define GTK 32

__global__ void k_gemm_bf16_nt(const bf16_t* __restrict__ A,
                               const bf16_t* __restrict__ B,
                               float* __restrict__ C,
                               int M, int N, int K) {
  __shared__ __align__(16) bf16_t As[2][GTM][GTK];   // 2 x 16 KB
  __shared__ __align__(16) bf16_t Bs[2][GTN][GTK];   // 2 x  2 KB

  const int t    = threadIdx.x;
  const int wave = t >> 5;
  const int lane = t & 31;
  const int row0 = blockIdx.x * GTM;
  const int col0 = blockIdx.y * GTN;

  const unsigned ldsA = lds_off(&As[0][0][0]);
  const unsigned ldsB = lds_off(&Bs[0][0][0]);

  auto stage = [&](int buf, int k0) {
#if HAVE_TDM
    if (wave == 0) {   // one TDM per tile; EXEC-independent DMA engine
      tdm_load_2d_bf16(ldsA + (unsigned)((buf * GTM * GTK) << 1),
                       A + (size_t)row0 * K + k0,
                       (unsigned)K, (unsigned)M, GTK, GTM, (unsigned)K);
      tdm_load_2d_bf16(ldsB + (unsigned)((buf * GTN * GTK) << 1),
                       B + (size_t)col0 * K + k0,
                       (unsigned)K, (unsigned)N, GTK, GTN, (unsigned)K);
    }
#else
    const bf16_t* Ag = A + (size_t)row0 * K + k0;
#pragma unroll
    for (int j = 0; j < 4; ++j) {
      int q  = t + j * 256;
      int r  = q >> 2;
      int co = (q & 3) * 8;
      async_to_lds_b128(ldsA + (unsigned)((buf * GTM * GTK + r * GTK + co) << 1),
                        Ag + (size_t)r * K + co);
    }
    if (t < 128) {
      int r  = t >> 2;
      int co = (t & 3) * 8;
      async_to_lds_b128(ldsB + (unsigned)((buf * GTN * GTK + r * GTK + co) << 1),
                        B + (size_t)(col0 + r) * K + k0 + co);
    }
#endif
  };

  v8f a00 = {}, a01 = {}, a10 = {}, a11 = {};

  stage(0, 0);
  wait_stage0();
  __syncthreads();

  int nb = 0;
  for (int k0 = 0; k0 < K; k0 += GTK) {
    if (k0 + GTK < K) stage(nb ^ 1, k0 + GTK);   // overlap next tile with compute

    const bf16_t* Ab = &As[nb][wave * 32][0];
    v16bf fa0 = load_frag(Ab, GTK, lane);
    v16bf fa1 = load_frag(Ab + 16 * GTK, GTK, lane);
    v16bf fb0 = load_frag(&Bs[nb][0][0], GTK, lane);
    v16bf fb1 = load_frag(&Bs[nb][16][0], GTK, lane);

    a00 = wmma_bf16(fa0, fb0, a00);
    a01 = wmma_bf16(fa0, fb1, a01);
    a10 = wmma_bf16(fa1, fb0, a10);
    a11 = wmma_bf16(fa1, fb1, a11);

    wait_stage0();        // next tile fully landed in LDS
    __syncthreads();      // publish / protect buffer reuse
    nb ^= 1;
  }

  const int n    = lane & 15;
  const int half = lane >> 4;
#pragma unroll
  for (int r = 0; r < 8; ++r) {
    int m0 = row0 + wave * 32 + r + half * 8;
    C[(size_t)m0 * N + col0 + n]             = a00[r];
    C[(size_t)m0 * N + col0 + 16 + n]        = a01[r];
    C[(size_t)(m0 + 16) * N + col0 + n]      = a10[r];
    C[(size_t)(m0 + 16) * N + col0 + 16 + n] = a11[r];
  }
}

// ---------------- depthwise causal conv (D_CONV=4) + bias + SiLU ----------------
__global__ void k_conv_silu(const float* __restrict__ zx,
                            const float* __restrict__ cw,
                            const float* __restrict__ cb,
                            float* __restrict__ xbc) {
  int i = blockIdx.x * 256 + threadIdx.x;
  if (i >= SEQ * CONVD) return;
  int l  = i / CONVD;
  int ch = i - l * CONVD;
  float s = cb[ch];
#pragma unroll
  for (int j = 0; j < 4; ++j) {
    int ls = l - 3 + j;
    if (ls >= 0) s += cw[ch * 4 + j] * zx[(size_t)ls * DIN + DI + ch];
  }
  xbc[i] = siluf(s);
}

// ---------------- dt = softplus(raw + bias); dA = dt * (-exp(A_log)) ----------------
__global__ void k_dt(const float* __restrict__ zx,
                     const float* __restrict__ dt_bias,
                     const float* __restrict__ A_log,
                     float* __restrict__ dt,
                     float* __restrict__ dA) {
  int i = blockIdx.x * 256 + threadIdx.x;
  if (i >= SEQ * NH) return;
  int l = i >> 5;
  int h = i & 31;
  float v  = zx[(size_t)l * DIN + (DIN - NH) + h] + dt_bias[h];
  float sp = (v > 20.0f) ? v : log1pf(__expf(v));
  dt[i] = sp;
  dA[i] = sp * (-__expf(A_log[h]));
}

// ---------------- bf16 re-layouts for SSD WMMA ----------------
__global__ void k_prep(const float* __restrict__ xbc,
                       bf16_t* __restrict__ xt,
                       bf16_t* __restrict__ Bb,
                       bf16_t* __restrict__ Cb) {
  int i = blockIdx.x * 256 + threadIdx.x;
  if (i >= SEQ * CONVD) return;
  int l  = i / CONVD;
  int ch = i - l * CONVD;
  float v = xbc[i];
  if (ch < DI) {
    int h = ch >> 7, p = ch & 127;
    int c = l >> 8, s = l & 255;
    xt[((((size_t)c * NH + h) * HD + p) << 8) + s] = (bf16_t)v;
  } else if (ch < DI + DS) {
    Bb[(size_t)l * DS + (ch - DI)] = (bf16_t)v;
  } else {
    Cb[(size_t)l * DS + (ch - DI - DS)] = (bf16_t)v;
  }
}

// ---------------- per-chunk inclusive cumsum of dA over l ----------------
__global__ void k_cumsum(const float* __restrict__ dA, float* __restrict__ Acum) {
  int c = blockIdx.x >> 5;
  int h = blockIdx.x & 31;
  int l = threadIdx.x;
  __shared__ float sbuf[CHK];
  sbuf[l] = dA[(size_t)(c * CHK + l) * NH + h];
  __syncthreads();
  for (int off = 1; off < CHK; off <<= 1) {
    float add = (l >= off) ? sbuf[l - off] : 0.0f;
    __syncthreads();
    sbuf[l] += add;
    __syncthreads();
  }
  Acum[((size_t)(c * NH + h) << 8) + l] = sbuf[l];
}

// ---------------- per-chunk states ----------------
__global__ void k_states(const float* __restrict__ xbc,
                         const float* __restrict__ Acum,
                         const float* __restrict__ dt,
                         float* __restrict__ states) {
  int c = blockIdx.x >> 5;
  int h = blockIdx.x & 31;
  int t = threadIdx.x;
  __shared__ float xs[32][HD];
  __shared__ float bs[32][DS];
  __shared__ float dec[32];

  const size_t base = (size_t)(c * NH + h) << 8;
  const float Alast = Acum[base + (CHK - 1)];
  const int p  = t >> 1;
  const int nb = (t & 1) * 32;

  float acc[32];
#pragma unroll
  for (int q = 0; q < 32; ++q) acc[q] = 0.0f;

  for (int lb = 0; lb < CHK; lb += 32) {
    for (int j = t; j < 32 * HD; j += 256) {
      int ll = j >> 7, pp = j & 127;
      xs[ll][pp] = xbc[(size_t)(c * CHK + lb + ll) * CONVD + h * HD + pp];
    }
    for (int j = t; j < 32 * DS; j += 256) {
      int ll = j >> 6, nn = j & 63;
      bs[ll][nn] = xbc[(size_t)(c * CHK + lb + ll) * CONVD + DI + nn];
    }
    if (t < 32) {
      dec[t] = __expf(Alast - Acum[base + lb + t]) *
               dt[(size_t)(c * CHK + lb + t) * NH + h];
    }
    __syncthreads();
    for (int ll = 0; ll < 32; ++ll) {
      float xv = xs[ll][p] * dec[ll];
#pragma unroll
      for (int q = 0; q < 32; ++q) acc[q] += xv * bs[ll][nb + q];
    }
    __syncthreads();
  }

  size_t ob = (((size_t)(c * NH + h) * HD + p) << 6) + nb;
#pragma unroll
  for (int q = 0; q < 32; ++q) states[ob + q] = acc[q];
}

// ---------------- inter-chunk recurrence ----------------
__global__ void k_recur(const float* __restrict__ states,
                        const float* __restrict__ Acum,
                        bf16_t* __restrict__ Stb) {
  int h = blockIdx.x;
  int t = threadIdx.x;
  float S[32];
#pragma unroll
  for (int q = 0; q < 32; ++q) S[q] = 0.0f;

  for (int c = 0; c < NC; ++c) {
    size_t ob = (((size_t)(c * NH + h)) << 13) + (size_t)t * 32;
#pragma unroll
    for (int q = 0; q < 32; ++q) Stb[ob + q] = (bf16_t)S[q];
    float ef = __expf(Acum[((size_t)(c * NH + h) << 8) + (CHK - 1)]);
#pragma unroll
    for (int q = 0; q < 32; ++q) S[q] = S[q] * ef + states[ob + q];
  }
}

// ---------------- SSD Y kernel (flash-style, all WMMA) ----------------
__global__ void k_ssd_y(const bf16_t* __restrict__ Cb,
                        const bf16_t* __restrict__ Bb,
                        const bf16_t* __restrict__ xt,
                        const bf16_t* __restrict__ Stb,
                        const float* __restrict__ Acum,
                        const float* __restrict__ dt,
                        const float* __restrict__ xbc,
                        const float* __restrict__ Dv,
                        float* __restrict__ y) {
  const int c = blockIdx.x >> 5;
  const int h = blockIdx.x & 31;
  const int t = threadIdx.x, wave = t >> 5, lane = t & 31;
  const int n = lane & 15, half = lane >> 4;

  __shared__ float As_[CHK];
  __shared__ float Ds_[CHK];
  __shared__ __align__(16) bf16_t G[8][16 * 32];

  As_[t] = Acum[((size_t)(c * NH + h) << 8) + t];
  Ds_[t] = dt[(size_t)(c * CHK + t) * NH + h];
  __syncthreads();

  const bf16_t* Cbase = Cb + (size_t)(c * CHK) * DS;
  const bf16_t* Bbase = Bb + (size_t)(c * CHK) * DS;
  const bf16_t* Xt    = xt + ((size_t)(c * NH + h) << 15);
  const bf16_t* St    = Stb + ((size_t)(c * NH + h) << 13);
  const float   Dh    = Dv[h];

  for (int it = wave; it < 16; it += 8) {
    const int L0 = it * 16;

    v16bf ca0 = load_frag(Cbase + (size_t)L0 * DS, DS, lane);
    v16bf ca1 = load_frag(Cbase + (size_t)L0 * DS + 32, DS, lane);

    v8f acc[8];
#pragma unroll
    for (int pt = 0; pt < 8; ++pt) acc[pt] = v8f{};
#pragma unroll
    for (int pt = 0; pt < 8; ++pt) {
      acc[pt] = wmma_bf16(ca0, load_frag(St + (size_t)(pt * 16) * DS,      DS, lane), acc[pt]);
      acc[pt] = wmma_bf16(ca1, load_frag(St + (size_t)(pt * 16) * DS + 32, DS, lane), acc[pt]);
    }
    float es[8];
#pragma unroll
    for (int r = 0; r < 8; ++r) es[r] = __expf(As_[L0 + r + half * 8]);
#pragma unroll
    for (int pt = 0; pt < 8; ++pt)
#pragma unroll
      for (int r = 0; r < 8; ++r) acc[pt][r] *= es[r];

    for (int sj = 0; sj <= (it >> 1); ++sj) {
#pragma unroll
      for (int u = 0; u < 2; ++u) {
        const int S0 = sj * 32 + u * 16;
        v8f g = {};
        g = wmma_bf16(ca0, load_frag(Bbase + (size_t)S0 * DS,      DS, lane), g);
        g = wmma_bf16(ca1, load_frag(Bbase + (size_t)S0 * DS + 32, DS, lane), g);
        const int   s_ = S0 + n;
        const float ds = Ds_[s_];
        const float as = As_[s_];
#pragma unroll
        for (int r = 0; r < 8; ++r) {
          const int l_ = L0 + r + half * 8;
          float f = (s_ <= l_) ? __expf(As_[l_] - as) * ds : 0.0f;
          G[wave][(r + half * 8) * 32 + u * 16 + n] = (bf16_t)(g[r] * f);
        }
      }
      v16bf ga = load_frag(&G[wave][0], 32, lane);
#pragma unroll
      for (int pt = 0; pt < 8; ++pt) {
        v16bf xb = load_frag(Xt + (size_t)(pt * 16) * CHK + sj * 32, CHK, lane);
        acc[pt] = wmma_bf16(ga, xb, acc[pt]);
      }
    }

#pragma unroll
    for (int pt = 0; pt < 8; ++pt) {
#pragma unroll
      for (int r = 0; r < 8; ++r) {
        const int l_ = L0 + r + half * 8;
        const int p_ = pt * 16 + n;
        const int gl = c * CHK + l_;
        float xv = xbc[(size_t)gl * CONVD + h * HD + p_];
        y[(size_t)gl * DI + h * HD + p_] = acc[pt][r] + Dh * xv;
      }
    }
  }
}

// ---------------- gated RMSNorm ----------------
__global__ void k_norm(const float* __restrict__ y,
                       const float* __restrict__ zx,
                       const float* __restrict__ nw,
                       bf16_t* __restrict__ gb) {
  const int l = blockIdx.x;
  const int t = threadIdx.x;
  __shared__ float red[256];
  float gv[16];
  float ss = 0.0f;
#pragma unroll
  for (int j = 0; j < 16; ++j) {
    int ch = t + j * 256;
    float yv = y[(size_t)l * DI + ch];
    float zv = zx[(size_t)l * DIN + ch];
    float g  = yv * siluf(zv);
    gv[j] = g;
    ss += g * g;
  }
  red[t] = ss;
  __syncthreads();
  for (int off = 128; off > 0; off >>= 1) {
    if (t < off) red[t] += red[t + off];
    __syncthreads();
  }
  float rs = rsqrtf(red[0] / (float)DI + 1e-5f);
#pragma unroll
  for (int j = 0; j < 16; ++j) {
    int ch = t + j * 256;
    gb[(size_t)l * DI + ch] = (bf16_t)(gv[j] * rs * nw[ch]);
  }
}

// ---------------- launcher ----------------
extern "C" void kernel_launch(void* const* d_in, const int* in_sizes, int n_in,
                              void* d_out, int out_size, void* d_ws, size_t ws_size,
                              hipStream_t stream) {
  const float* x        = (const float*)d_in[0];
  const float* in_proj  = (const float*)d_in[1];
  const float* conv_w   = (const float*)d_in[2];
  const float* conv_b   = (const float*)d_in[3];
  const float* dt_bias  = (const float*)d_in[4];
  const float* A_log    = (const float*)d_in[5];
  const float* Dv       = (const float*)d_in[6];
  const float* norm_w   = (const float*)d_in[7];
  const float* out_proj = (const float*)d_in[8];
  float* out = (float*)d_out;

  char* ws = (char*)d_ws;
  size_t o = 0;
  auto alloc = [&](size_t bytes) { size_t r = o; o += (bytes + 255) & ~(size_t)255; return r; };

  bf16_t* xb     = (bf16_t*)(ws + alloc((size_t)SEQ * DIM * 2));
  bf16_t* wpb    = (bf16_t*)(ws + alloc((size_t)DIN * DIM * 2));
  bf16_t* wob    = (bf16_t*)(ws + alloc((size_t)DIM * DI * 2));
  float*  zx     = (float*)(ws + alloc((size_t)SEQ * DIN * 4));
  float*  xbc    = (float*)(ws + alloc((size_t)SEQ * CONVD * 4));
  float*  dt     = (float*)(ws + alloc((size_t)SEQ * NH * 4));
  float*  dA     = (float*)(ws + alloc((size_t)SEQ * NH * 4));
  float*  Acum   = (float*)(ws + alloc((size_t)NC * NH * CHK * 4));
  bf16_t* Bb     = (bf16_t*)(ws + alloc((size_t)SEQ * DS * 2));
  bf16_t* Cbuf   = (bf16_t*)(ws + alloc((size_t)SEQ * DS * 2));
  bf16_t* xt     = (bf16_t*)(ws + alloc((size_t)NC * NH * HD * CHK * 2));
  float*  states = (float*)(ws + alloc((size_t)NC * NH * HD * DS * 4));
  bf16_t* Stb    = (bf16_t*)(ws + alloc((size_t)NC * NH * HD * DS * 2));
  float*  ybuf   = (float*)(ws + alloc((size_t)SEQ * DI * 4));
  bf16_t* gb     = (bf16_t*)(ws + alloc((size_t)SEQ * DI * 2));

  k_cvt_bf16<<<(SEQ * DIM + 255) / 256, 256, 0, stream>>>(x, xb, SEQ * DIM);
  k_cvt_bf16<<<(DIN * DIM + 255) / 256, 256, 0, stream>>>(in_proj, wpb, DIN * DIM);
  k_cvt_bf16<<<(DIM * DI + 255) / 256, 256, 0, stream>>>(out_proj, wob, DIM * DI);

  k_gemm_bf16_nt<<<dim3(SEQ / GTM, DIN / GTN), 256, 0, stream>>>(xb, wpb, zx, SEQ, DIN, DIM);

  k_conv_silu<<<(SEQ * CONVD + 255) / 256, 256, 0, stream>>>(zx, conv_w, conv_b, xbc);
  k_dt<<<(SEQ * NH + 255) / 256, 256, 0, stream>>>(zx, dt_bias, A_log, dt, dA);
  k_prep<<<(SEQ * CONVD + 255) / 256, 256, 0, stream>>>(xbc, xt, Bb, Cbuf);

  k_cumsum<<<NC * NH, CHK, 0, stream>>>(dA, Acum);
  k_states<<<NC * NH, 256, 0, stream>>>(xbc, Acum, dt, states);
  k_recur<<<NH, 256, 0, stream>>>(states, Acum, Stb);

  k_ssd_y<<<NC * NH, 256, 0, stream>>>(Cbuf, Bb, xt, Stb, Acum, dt, xbc, Dv, ybuf);

  k_norm<<<SEQ, 256, 0, stream>>>(ybuf, zx, norm_w, gb);

  k_gemm_bf16_nt<<<dim3(SEQ / GTM, DIM / GTN), 256, 0, stream>>>(gb, wob, out, SEQ, DIM, DI);

  (void)in_sizes; (void)n_in; (void)out_size; (void)ws_size;
}
